// _NonLocalBlock_32427003084831
// MI455X (gfx1250) — matrix operand order repaired and checked by reading, not compiled, not run on hardware
//
#include <hip/hip_runtime.h>

typedef __attribute__((ext_vector_type(16))) _Float16 v16h;
typedef __attribute__((ext_vector_type(8)))  float    v8f;
typedef __attribute__((ext_vector_type(4)))  unsigned v4u;

#define Bsz 8
#define Cch 128
#define Nn  2048
#define ICc 64
#define NEGV (-9000000000000000.0f)

// ws layout (float units): ct[128] | cp[128] | bt,bp @256 | ftW[16384] @512 | fpW[16384] @16896
// then gt (f16, [b][o][n], 8*64*2048) at float offset 33280 (byte offset 133120, 16B-aligned)

__global__ void k_combine(const float* __restrict__ theta_w, const float* __restrict__ phi_w,
                          const float* __restrict__ theta_b, const float* __restrict__ phi_b,
                          const float* __restrict__ cp_w, float* __restrict__ ws) {
    int c = threadIdx.x;  // 0..127
    float a = 0.f, p = 0.f;
    for (int o = 0; o < ICc; ++o) {
        a += cp_w[o]       * theta_w[o * Cch + c];
        p += cp_w[ICc + o] * phi_w[o * Cch + c];
    }
    ws[c]       = a;
    ws[128 + c] = p;
    if (c == 0) { float bt = 0.f; for (int o = 0; o < ICc; ++o) bt += cp_w[o] * theta_b[o];       ws[256] = bt; }
    if (c == 1) { float bp = 0.f; for (int o = 0; o < ICc; ++o) bp += cp_w[ICc + o] * phi_b[o];   ws[257] = bp; }
}

__global__ void k_fvec(const float* __restrict__ x, const float* __restrict__ ws,
                       float* __restrict__ ftW, float* __restrict__ fpW) {
    int id = blockIdx.x * blockDim.x + threadIdx.x;   // over B*N
    int b = id >> 11, n = id & 2047;
    const float* xp = x + (((size_t)b * Cch) << 11) + n;
    float ft = ws[256], fp = ws[257];
    for (int c = 0; c < Cch; ++c) {
        float xv = xp[(size_t)c << 11];
        ft += ws[c] * xv;
        fp += ws[128 + c] * xv;
    }
    ftW[id] = ft;
    fpW[id] = fp;
}

__global__ void k_gx(const float* __restrict__ x, const float* __restrict__ g_w,
                     const float* __restrict__ g_b, _Float16* __restrict__ gt) {
    int id = blockIdx.x * blockDim.x + threadIdx.x;   // over B*IC*N, layout [b][o][n]
    int b = id >> 17;
    int o = (id >> 11) & 63;
    int n = id & 2047;
    const float* xp = x + (((size_t)b * Cch) << 11) + n;
    const float* gw = g_w + o * Cch;
    float acc = g_b[o];
    for (int c = 0; c < Cch; ++c) acc += gw[c] * xp[(size_t)c << 11];
    gt[id] = (_Float16)acc;
}

__global__ __launch_bounds__(256) void k_attn(const int* __restrict__ adj,
                                              const float* __restrict__ ftW,
                                              const float* __restrict__ fpW,
                                              const _Float16* __restrict__ gt,
                                              float* __restrict__ out) {
    __shared__ float    fpS[2048];
    __shared__ float    ftS[16];
    __shared__ float    rowmaxS[16];
    __shared__ float    rowsumS[16];
    __shared__ unsigned maskS[16 * 64];
    __shared__ float    cS[8 * 1024];

    const int tid  = threadIdx.x;
    const int lane = tid & 31;
    const int w    = tid >> 5;        // wave 0..7
    const int half = lane >> 4;
    const int Ml   = lane & 15;
    const int b    = blockIdx.x >> 7;
    const int i0   = (blockIdx.x & 127) << 4;

    for (int k = tid; k < 2048; k += 256) fpS[k] = fpW[(b << 11) + k];
    if (tid < 16) ftS[tid] = ftW[(b << 11) + i0 + tid];
    __syncthreads();

    // ---- pass 1: per-row max, sum(exp), packed adj bitmask (2 rows per wave, coalesced)
    for (int rr = 0; rr < 2; ++rr) {
        int r = 2 * w + rr;
        const int* adjRow = adj + (size_t)(i0 + r) * 2048;
        float fti = ftS[r];
        float m = -3.0e38f;
        for (int jb = 0; jb < 2048; jb += 32) {
            int j = jb + lane;
            int av = adjRow[j];
            float s = fti + fpS[j];
            s = (s > 0.f) ? s : 0.2f * s;
            bool msk = av > 0;
            float e = msk ? NEGV : s;
            m = fmaxf(m, e);
            unsigned bit = msk ? (1u << lane) : 0u;
            #pragma unroll
            for (int off = 16; off; off >>= 1) bit |= __shfl_xor(bit, off);
            if (lane == 0) maskS[r * 64 + (jb >> 5)] = bit;
        }
        #pragma unroll
        for (int off = 16; off; off >>= 1) m = fmaxf(m, __shfl_xor(m, off));
        float sum = 0.f;
        for (int jb = 0; jb < 2048; jb += 32) {
            int j = jb + lane;
            unsigned mw = maskS[r * 64 + (jb >> 5)];
            float s = fti + fpS[j];
            s = (s > 0.f) ? s : 0.2f * s;
            float e = ((mw >> lane) & 1u) ? NEGV : s;
            sum += __expf(e - m);
        }
        #pragma unroll
        for (int off = 16; off; off >>= 1) sum += __shfl_xor(sum, off);
        if (lane == 0) { rowmaxS[r] = m; rowsumS[r] = sum; }
    }
    __syncthreads();

    // ---- pass 2: double-buffered B tiles; loads for chunk c+8 fly under chunk c's exps+WMMAs
    const float ftv = ftS[Ml];
    const float rm  = rowmaxS[Ml];
    v8f acc[4];
    #pragma unroll
    for (int t = 0; t < 4; ++t) { v8f z = {}; acc[t] = z; }

    union AU { _Float16 h[16]; v16h v; };
    union BU { v4u q[2]; v16h v; };
    const unsigned* gtd = (const unsigned*)gt;

    // per-lane dword base for each otile's gt row (constant across chunks)
    size_t bbase[4];
    #pragma unroll
    for (int t = 0; t < 4; ++t)
        bbase[t] = ((((size_t)(b * 64 + t * 16 + Ml)) << 11) >> 1) + (size_t)(half * 8);

    auto loadB = [&](BU* bm, int chunk) {
        const int jb = chunk << 5;
        #pragma unroll
        for (int t = 0; t < 4; ++t) {
            const v4u* p = (const v4u*)(gtd + bbase[t] + (jb >> 1));
            bm[t].q[0] = p[0];
            bm[t].q[1] = p[1];
        }
    };

    auto stepChunk = [&](const BU* bm, int chunk) {
        const int jb = chunk << 5;
        const unsigned mw = maskS[Ml * 64 + chunk];
        AU a;
        #pragma unroll
        for (int v = 0; v < 8; ++v) {
            int kb = half * 16 + 2 * v;
            #pragma unroll
            for (int q = 0; q < 2; ++q) {
                int k = kb + q;
                float s = ftv + fpS[jb + k];
                s = (s > 0.f) ? s : 0.2f * s;
                float e = ((mw >> k) & 1u) ? NEGV : s;
                a.h[2 * v + q] = (_Float16)__expf(e - rm);
            }
        }
        #pragma unroll
        for (int t = 0; t < 4; ++t)
            acc[t] = __builtin_amdgcn_wmma_f32_16x16x32_f16(
                false, a.v, false, bm[t].v, (short)0, acc[t], false, false);
    };

    BU bmA[4], bmB[4];
    loadB(bmA, w);                       // chunks owned by this wave: w, w+8, ..., w+56
    for (int chunk = w; chunk < 64; chunk += 16) {
        loadB(bmB, chunk + 8);           // chunk+8 <= 63 always
        stepChunk(bmA, chunk);
        if (chunk + 16 < 64) loadB(bmA, chunk + 16);
        stepChunk(bmB, chunk + 8);
    }

    // ---- cross-wave reduction of partial C tiles, apply 1/rowsum, write [b][o][n]
    #pragma unroll
    for (int t = 0; t < 4; ++t) {
        #pragma unroll
        for (int g = 0; g < 8; ++g) {
            int Mc = g + half * 8;
            cS[w * 1024 + Mc * 64 + t * 16 + Ml] = acc[t][g];
        }
    }
    __syncthreads();

    #pragma unroll
    for (int k = 0; k < 4; ++k) {
        int q = tid + k * 256;
        int Mq = q >> 6;
        int o  = q & 63;
        float s = 0.f;
        #pragma unroll
        for (int wv = 0; wv < 8; ++wv) s += cS[wv * 1024 + q];
        s /= rowsumS[Mq];
        out[(((size_t)(b * 64 + o)) << 11) + i0 + Mq] = s;
    }
}

extern "C" void kernel_launch(void* const* d_in, const int* in_sizes, int n_in,
                              void* d_out, int out_size, void* d_ws, size_t ws_size,
                              hipStream_t stream) {
    const float* x       = (const float*)d_in[0];
    const int*   adj     = (const int*)d_in[1];
    const float* g_w     = (const float*)d_in[2];
    const float* g_b     = (const float*)d_in[3];
    const float* theta_w = (const float*)d_in[4];
    const float* theta_b = (const float*)d_in[5];
    const float* phi_w   = (const float*)d_in[6];
    const float* phi_b   = (const float*)d_in[7];
    const float* cp_w    = (const float*)d_in[8];

    float*     ws  = (float*)d_ws;
    float*     ftW = ws + 512;
    float*     fpW = ws + 512 + 16384;
    _Float16*  gt  = (_Float16*)(ws + 512 + 2 * 16384);
    float*     out = (float*)d_out;

    k_combine<<<1, 128, 0, stream>>>(theta_w, phi_w, theta_b, phi_b, cp_w, ws);
    k_fvec<<<(Bsz * Nn) / 256, 256, 0, stream>>>(x, ws, ftW, fpW);
    k_gx<<<(Bsz * ICc * Nn) / 256, 256, 0, stream>>>(x, g_w, g_b, gt);
    k_attn<<<Bsz * (Nn / 16), 256, 0, stream>>>(adj, ftW, fpW, gt, out);
}
